// BlockSparseMoE_10574209483532
// MI455X (gfx1250) — compile-verified, matrix operand order checked by reading, not run on hardware
//
#include <hip/hip_runtime.h>

// ---------------------------------------------------------------------------
// BlockSparse MoE (T=8192, D=2048, F=3584, E=8, top_k=2) for gfx1250 (CDNA5).
// Routed top-2. GEMMs on v_wmma_f32_16x16x32_bf16.
//  - B operands: 64-K stages async-copied to LDS (GLOBAL_LOAD_ASYNC_TO_LDS_B128,
//    ASYNCcnt, double-buffered, one stage ahead).
//  - A operands: register double-buffer, one stage ahead; stage loop unrolled
//    x2 so buffer rotation is pure register renaming (no moves, no hazard NOPs).
// ---------------------------------------------------------------------------

static constexpr int kT = 8192;   // tokens
static constexpr int kD = 2048;   // model dim
static constexpr int kF = 3584;   // ffn dim
static constexpr int kE = 8;      // experts
static constexpr int kTwoF = 2 * kF;
static constexpr int kRows = 2 * kT;  // routed (token,expert) pairs

// LDS row pitch: 64 K-elems (128B) + 16B pad = 144B (multiple of 16B; stride
// 36 banks, gcd(36,64)=4 -> 16 distinct 4-bank groups across lanes).
static constexpr int kPitchS = 72;            // shorts per row
static constexpr int kPitchB = kPitchS * 2;   // bytes per row

typedef __attribute__((ext_vector_type(16))) __bf16 v16bf;
typedef __attribute__((ext_vector_type(8)))  float  v8f;
typedef __attribute__((ext_vector_type(8)))  unsigned short v8us;

union FragBF {
  v16bf bf;
  v8us  h[2];
  unsigned short us[16];
};

__device__ __forceinline__ unsigned short f2bf(float f) {
  unsigned u = __float_as_uint(f);
  u += 0x7FFFu + ((u >> 16) & 1u);   // round-to-nearest-even
  return (unsigned short)(u >> 16);
}

__device__ __forceinline__ float silu(float v) {
  return v / (1.0f + __expf(-v));
}

// A/B 16x32 bf16 fragment: lane<16 holds K {0..7,16..23}, lane>=16 {8..15,24..31}.
__device__ __forceinline__ v16bf load_frag(const unsigned short* row_k0, int lane) {
  const int koff = (lane & 16) ? 8 : 0;
  FragBF f;
  f.h[0] = *reinterpret_cast<const v8us*>(row_k0 + koff);
  f.h[1] = *reinterpret_cast<const v8us*>(row_k0 + 16 + koff);
  return f.bf;
}

__device__ __forceinline__ v8f wmma(v16bf a, v16bf b, v8f c) {
  return __builtin_amdgcn_wmma_f32_16x16x32_bf16(false, a, false, b, (short)0, c,
                                                 false, false);
}
// Same B operand as the (identical) previous WMMA -> set reuse_b hint.
__device__ __forceinline__ v8f wmma_rb(v16bf a, v16bf b, v8f c) {
  return __builtin_amdgcn_wmma_f32_16x16x32_bf16(false, a, false, b, (short)0, c,
                                                 false, true);
}

// Async copy 16B global -> LDS (tracked by ASYNCcnt, not LOADcnt).
__device__ __forceinline__ void async_b128(unsigned lds_byte_addr, const void* gaddr) {
  asm volatile("global_load_async_to_lds_b128 %0, %1, off"
               :: "v"(lds_byte_addr), "v"(gaddr) : "memory");
}

// ---------------------------------------------------------------------------
// Kernel 1: fp32 -> packed bf16 (2 floats -> 1 dword), grid-stride, mem-bound.
// ---------------------------------------------------------------------------
__global__ void k_cvt(const float* __restrict__ src,
                      unsigned int* __restrict__ dst, int npairs) {
  int i = blockIdx.x * blockDim.x + threadIdx.x;
  int stride = gridDim.x * blockDim.x;
  for (; i < npairs; i += stride) {
    unsigned lo = f2bf(src[2 * i]);
    unsigned hi = f2bf(src[2 * i + 1]);
    dst[i] = lo | (hi << 16);
  }
}

// ---------------------------------------------------------------------------
// Kernel 2: router — one wave per token.
// ---------------------------------------------------------------------------
__global__ __launch_bounds__(256) void k_router(const float* __restrict__ x,
                                                const float* __restrict__ gate_w,
                                                int* __restrict__ sel,
                                                float* __restrict__ wts) {
  const int wave = threadIdx.x >> 5;
  const int lane = threadIdx.x & 31;
  const int t = blockIdx.x * 8 + wave;
  if (t >= kT) return;

  float acc[kE];
#pragma unroll
  for (int e = 0; e < kE; ++e) acc[e] = 0.0f;

  const float* xr = x + (size_t)t * kD;
  for (int k = lane; k < kD; k += 32) {
    float xv = xr[k];
#pragma unroll
    for (int e = 0; e < kE; ++e) acc[e] += xv * gate_w[e * kD + k];
  }
#pragma unroll
  for (int e = 0; e < kE; ++e) {
#pragma unroll
    for (int off = 16; off > 0; off >>= 1)
      acc[e] += __shfl_xor(acc[e], off, 32);
  }

  if (lane == 0) {
    float m = acc[0];
#pragma unroll
    for (int e = 1; e < kE; ++e) m = fmaxf(m, acc[e]);
    float p[kE];
#pragma unroll
    for (int e = 0; e < kE; ++e) p[e] = __expf(acc[e] - m);
    int i0 = 0;
#pragma unroll
    for (int e = 1; e < kE; ++e) if (p[e] > p[i0]) i0 = e;
    int i1 = (i0 == 0) ? 1 : 0;
#pragma unroll
    for (int e = 0; e < kE; ++e)
      if (e != i0 && p[e] > p[i1]) i1 = e;
    float s = p[i0] + p[i1];
    sel[2 * t]     = i0;
    sel[2 * t + 1] = i1;
    wts[2 * t]     = p[i0] / s;
    wts[2 * t + 1] = p[i1] / s;
  }
}

// ---------------------------------------------------------------------------
// Kernel 3: deterministic per-expert compaction (1 block; wave e = expert e)
// ---------------------------------------------------------------------------
__global__ __launch_bounds__(256) void k_build(const int* __restrict__ sel,
                                               const float* __restrict__ wts,
                                               int* __restrict__ token_ids,
                                               float* __restrict__ token_wt,
                                               int* __restrict__ counts,
                                               int* __restrict__ offs) {
  __shared__ int sc[kE], so[kE];
  const int e = threadIdx.x >> 5;
  const int lane = threadIdx.x & 31;

  int cnt = 0;
  for (int t = lane; t < kT; t += 32) {
    bool m = (sel[2 * t] == e) || (sel[2 * t + 1] == e);
    unsigned long long bal = __ballot(m);
    if (lane == 0) cnt += __popcll(bal);
  }
  if (lane == 0) sc[e] = cnt;
  __syncthreads();
  if (threadIdx.x == 0) {
    int r = 0;
    for (int i = 0; i < kE; ++i) { so[i] = r; r += sc[i]; }
    for (int i = 0; i < kE; ++i) { counts[i] = sc[i]; offs[i] = so[i]; }
  }
  __syncthreads();

  int pos = so[e];
  for (int t = lane; t < kT; t += 32) {
    int s0 = sel[2 * t], s1 = sel[2 * t + 1];
    bool m = (s0 == e) || (s1 == e);
    float w = (s0 == e) ? wts[2 * t] : wts[2 * t + 1];
    unsigned long long bal = __ballot(m);
    int before = __popcll(bal & ((1ull << lane) - 1ull));
    if (m) {
      token_ids[pos + before] = t;
      token_wt[pos + before]  = w;
    }
    pos += __popcll(bal);
  }
}

// ---------------------------------------------------------------------------
// Kernel 4: GEMM1 + fused SiLU*up.  Macro tile 128(M) x 64(N gate + paired up).
// 64-K stages: B (gate 64 + up 64 rows) async to LDS; A register-pipelined.
// 16 WMMAs per stage per wave.
// ---------------------------------------------------------------------------
__global__ __launch_bounds__(256) void k_gemm1(const unsigned short* __restrict__ xb,
                                               const unsigned short* __restrict__ wv1b,
                                               const int* __restrict__ token_ids,
                                               const int* __restrict__ counts,
                                               const int* __restrict__ offs,
                                               unsigned short* __restrict__ act) {
  // 2 buffers x 128 rows x 72 shorts (144B pitch) = 36.0 KB
  __shared__ __align__(16) unsigned short smem[2 * 128 * kPitchS];

  const int e = blockIdx.z;
  const int cnt = counts[e];
  const int m0 = blockIdx.y * 128;
  if (m0 >= cnt) return;
  const int n0 = blockIdx.x * 64;
  const int off = offs[e];

  const int wave = threadIdx.x >> 5;
  const int lane = threadIdx.x & 31;
  const int mi = wave & 3;        // 32-row strip of the 128-row macro tile
  const int nh = wave >> 2;       // 32-col strip of the 64-col macro tile

  // --- async staging: thread -> (B row j2 of 128, 4 of 8 16B chunks) ---
  const int j2 = threadIdx.x >> 1;          // 0..127 (0..63 gate, 64..127 up)
  const int cg4 = (threadIdx.x & 1) * 4;    // first chunk: 0 or 4
  const unsigned short* grow =
      (j2 < 64) ? wv1b + ((size_t)e * kTwoF + n0 + j2) * kD
                : wv1b + ((size_t)e * kTwoF + kF + n0 + (j2 - 64)) * kD;
  const unsigned ldsb = (unsigned)(unsigned long long)&smem[0];
  const unsigned myl = ldsb + (unsigned)(j2 * kPitchB + cg4 * 16);
  const unsigned bufBytes = 128 * kPitchB;

  auto stage = [&](int buf, int kbase) {
    const unsigned short* g = grow + kbase + cg4 * 8;
    unsigned l = myl + (unsigned)buf * bufBytes;
#pragma unroll
    for (int i = 0; i < 4; ++i) async_b128(l + i * 16, g + i * 8);
  };

  // --- A: two gathered rows per lane, register double-buffered per stage ---
  int ra0 = m0 + mi * 32 + (lane & 15);
  int ra1 = ra0 + 16;
  int rc0 = (ra0 < cnt) ? ra0 : (cnt - 1);
  int rc1 = (ra1 < cnt) ? ra1 : (cnt - 1);
  const unsigned short* ap0 = xb + (size_t)token_ids[off + rc0] * kD;
  const unsigned short* ap1 = xb + (size_t)token_ids[off + rc1] * kD;

  const int lrow = nh * 32 + (lane & 15);   // LDS row for this lane's B column

  v8f cg00 = {}, cg01 = {}, cg10 = {}, cg11 = {};
  v8f cu00 = {}, cu01 = {}, cu10 = {}, cu11 = {};

  // prologue: stage 0 async + A regs for stage 0
  stage(0, 0);
  v16bf a0c0 = load_frag(ap0, lane),      a1c0 = load_frag(ap1, lane);
  v16bf a0c1 = load_frag(ap0 + 32, lane), a1c1 = load_frag(ap1 + 32, lane);

  constexpr int kStages = kD / 64;   // 32 (even -> unroll 2 divides exactly)
  int buf = 0;
#pragma unroll 2
  for (int s = 0; s < kStages; ++s, buf ^= 1) {
    const int kbase = s * 64;
    const bool more = (s + 1 < kStages);
    if (more) {
      stage(buf ^ 1, kbase + 64);
      asm volatile("s_wait_asynccnt 4" ::: "memory");  // current stage landed
    } else {
      asm volatile("s_wait_asynccnt 0" ::: "memory");
    }
    __syncthreads();

    // prefetch A for next stage (clamped on last stage; uniform branch-free)
    const int kn = more ? kbase + 64 : 0;
    v16bf a0n0 = load_frag(ap0 + kn, lane),      a1n0 = load_frag(ap1 + kn, lane);
    v16bf a0n1 = load_frag(ap0 + kn + 32, lane), a1n1 = load_frag(ap1 + kn + 32, lane);

    const unsigned short* sb = smem + buf * (128 * kPitchS);
#pragma unroll
    for (int kk = 0; kk < 2; ++kk) {
      const v16bf a0 = kk ? a0c1 : a0c0;
      const v16bf a1 = kk ? a1c1 : a1c0;
      const int ko = kk * 32;
      v16bf b;
      b = load_frag(sb + lrow * kPitchS + ko, lane);              // gate, n=0
      cg00 = wmma(a0, b, cg00);  cg10 = wmma_rb(a1, b, cg10);
      b = load_frag(sb + (lrow + 16) * kPitchS + ko, lane);       // gate, n=1
      cg01 = wmma(a0, b, cg01);  cg11 = wmma_rb(a1, b, cg11);
      b = load_frag(sb + (64 + lrow) * kPitchS + ko, lane);       // up, n=0
      cu00 = wmma(a0, b, cu00);  cu10 = wmma_rb(a1, b, cu10);
      b = load_frag(sb + (64 + lrow + 16) * kPitchS + ko, lane);  // up, n=1
      cu01 = wmma(a0, b, cu01);  cu11 = wmma_rb(a1, b, cu11);
    }
    a0c0 = a0n0; a1c0 = a1n0; a0c1 = a0n1; a1c1 = a1n1;

    __syncthreads();  // reads done before next stage overwrites this buffer
  }

  const int hi8 = (lane & 16) ? 8 : 0;
  const int cb = n0 + nh * 32 + (lane & 15);
  const v8f* cg[2][2] = {{&cg00, &cg01}, {&cg10, &cg11}};
  const v8f* cu[2][2] = {{&cu00, &cu01}, {&cu10, &cu11}};
#pragma unroll
  for (int m = 0; m < 2; ++m) {
#pragma unroll
    for (int i = 0; i < 8; ++i) {
      int rr = m0 + mi * 32 + m * 16 + i + hi8;
      if (rr < cnt) {
        size_t rowoff = (size_t)(off + rr) * kF;
#pragma unroll
        for (int n = 0; n < 2; ++n) {
          float g = (*cg[m][n])[i];
          float u = (*cu[m][n])[i];
          act[rowoff + cb + n * 16] = f2bf(silu(g) * u);
        }
      }
    }
  }
}

// ---------------------------------------------------------------------------
// Kernel 5: GEMM2. y = act @ w2[e].T ; out[token] += w_token * y (2 atomics/elt
// onto zero -> deterministic). 64-K stages; B async to LDS; A reg-pipelined.
// ---------------------------------------------------------------------------
__global__ __launch_bounds__(256) void k_gemm2(const unsigned short* __restrict__ act,
                                               const unsigned short* __restrict__ w2b,
                                               const int* __restrict__ token_ids,
                                               const float* __restrict__ token_wt,
                                               const int* __restrict__ counts,
                                               const int* __restrict__ offs,
                                               float* __restrict__ out) {
  // 2 buffers x 64 rows x 72 shorts = 18.0 KB
  __shared__ __align__(16) unsigned short smem[2 * 64 * kPitchS];

  const int e = blockIdx.z;
  const int cnt = counts[e];
  const int m0 = blockIdx.y * 128;
  if (m0 >= cnt) return;
  const int n0 = blockIdx.x * 64;
  const int off = offs[e];

  const int wave = threadIdx.x >> 5;
  const int lane = threadIdx.x & 31;
  const int mi = wave & 3;
  const int nh = wave >> 2;

  // staging: thread -> (B row j of 64, 2 of 8 chunks)
  const int j = threadIdx.x >> 2;           // 0..63
  const int cg2 = (threadIdx.x & 3) * 2;    // first chunk: 0,2,4,6
  const unsigned short* grow = w2b + ((size_t)e * kD + n0 + j) * kF;  // [E][D][F]
  const unsigned ldsb = (unsigned)(unsigned long long)&smem[0];
  const unsigned myl = ldsb + (unsigned)(j * kPitchB + cg2 * 16);
  const unsigned bufBytes = 64 * kPitchB;

  auto stage = [&](int buf, int kbase) {
    const unsigned short* g = grow + kbase + cg2 * 8;
    unsigned l = myl + (unsigned)buf * bufBytes;
#pragma unroll
    for (int i = 0; i < 2; ++i) async_b128(l + i * 16, g + i * 8);
  };

  int ra0 = m0 + mi * 32 + (lane & 15);
  int ra1 = ra0 + 16;
  int rc0 = (ra0 < cnt) ? ra0 : (cnt - 1);
  int rc1 = (ra1 < cnt) ? ra1 : (cnt - 1);
  const unsigned short* ap0 = act + (size_t)(off + rc0) * kF;
  const unsigned short* ap1 = act + (size_t)(off + rc1) * kF;

  const int lrow = nh * 32 + (lane & 15);

  v8f c00 = {}, c01 = {}, c10 = {}, c11 = {};

  stage(0, 0);
  v16bf a0c0 = load_frag(ap0, lane),      a1c0 = load_frag(ap1, lane);
  v16bf a0c1 = load_frag(ap0 + 32, lane), a1c1 = load_frag(ap1 + 32, lane);

  constexpr int kStages = kF / 64;   // 56 (even -> unroll 2 divides exactly)
  int buf = 0;
#pragma unroll 2
  for (int s = 0; s < kStages; ++s, buf ^= 1) {
    const int kbase = s * 64;
    const bool more = (s + 1 < kStages);
    if (more) {
      stage(buf ^ 1, kbase + 64);
      asm volatile("s_wait_asynccnt 2" ::: "memory");
    } else {
      asm volatile("s_wait_asynccnt 0" ::: "memory");
    }
    __syncthreads();

    const int kn = more ? kbase + 64 : 0;
    v16bf a0n0 = load_frag(ap0 + kn, lane),      a1n0 = load_frag(ap1 + kn, lane);
    v16bf a0n1 = load_frag(ap0 + kn + 32, lane), a1n1 = load_frag(ap1 + kn + 32, lane);

    const unsigned short* sb = smem + buf * (64 * kPitchS);
#pragma unroll
    for (int kk = 0; kk < 2; ++kk) {
      const v16bf a0 = kk ? a0c1 : a0c0;
      const v16bf a1 = kk ? a1c1 : a1c0;
      const int ko = kk * 32;
      v16bf b;
      b = load_frag(sb + lrow * kPitchS + ko, lane);
      c00 = wmma(a0, b, c00);  c10 = wmma_rb(a1, b, c10);
      b = load_frag(sb + (lrow + 16) * kPitchS + ko, lane);
      c01 = wmma(a0, b, c01);  c11 = wmma_rb(a1, b, c11);
    }
    a0c0 = a0n0; a1c0 = a1n0; a0c1 = a0n1; a1c1 = a1n1;

    __syncthreads();
  }

  const int hi8 = (lane & 16) ? 8 : 0;
  const int cb = n0 + nh * 32 + (lane & 15);
  const v8f* cc[2][2] = {{&c00, &c01}, {&c10, &c11}};
#pragma unroll
  for (int m = 0; m < 2; ++m) {
#pragma unroll
    for (int i = 0; i < 8; ++i) {
      int rr = m0 + mi * 32 + m * 16 + i + hi8;
      if (rr < cnt) {
        int p = off + rr;
        int t = token_ids[p];
        float w = token_wt[p];
#pragma unroll
        for (int n = 0; n < 2; ++n) {
          atomicAdd(out + (size_t)t * kD + cb + n * 16, w * (*cc[m][n])[i]);
        }
      }
    }
  }
}

// ---------------------------------------------------------------------------
// Host-side launcher
// ---------------------------------------------------------------------------
extern "C" void kernel_launch(void* const* d_in, const int* in_sizes, int n_in,
                              void* d_out, int out_size, void* d_ws, size_t ws_size,
                              hipStream_t stream) {
  (void)in_sizes; (void)n_in; (void)out_size; (void)ws_size;

  const float* x      = (const float*)d_in[0];
  const float* gate_w = (const float*)d_in[1];
  const float* wv1    = (const float*)d_in[2];
  const float* w2     = (const float*)d_in[3];
  float* out          = (float*)d_out;

  char* ws = (char*)d_ws;
  size_t o = 0;
  auto carve = [&](size_t bytes) {
    size_t cur = o;
    o += (bytes + 255) & ~(size_t)255;
    return cur;
  };
  unsigned short* xb   = (unsigned short*)(ws + carve((size_t)kT * kD * 2));
  unsigned short* wv1b = (unsigned short*)(ws + carve((size_t)kE * kTwoF * kD * 2));
  unsigned short* w2b  = (unsigned short*)(ws + carve((size_t)kE * kD * kF * 2));
  unsigned short* act  = (unsigned short*)(ws + carve((size_t)kRows * kF * 2));
  int*   sel           = (int*)  (ws + carve((size_t)kRows * 4));
  float* wts           = (float*)(ws + carve((size_t)kRows * 4));
  int*   token_ids     = (int*)  (ws + carve((size_t)kRows * 4));
  float* token_wt      = (float*)(ws + carve((size_t)kRows * 4));
  int*   counts        = (int*)  (ws + carve((size_t)kE * 4));
  int*   offs          = (int*)  (ws + carve((size_t)kE * 4));

  hipMemsetAsync(d_out, 0, (size_t)kT * kD * sizeof(float), stream);

  // bf16 conversions (memory-bound, ~1 GB traffic ≈ 45 us at 23.3 TB/s)
  k_cvt<<<4096, 256, 0, stream>>>(x,   (unsigned int*)xb,   kT * kD / 2);
  k_cvt<<<8192, 256, 0, stream>>>(wv1, (unsigned int*)wv1b, kE * kTwoF * kD / 2);
  k_cvt<<<8192, 256, 0, stream>>>(w2,  (unsigned int*)w2b,  kE * kD * kF / 2);

  k_router<<<kT / 8, 256, 0, stream>>>(x, gate_w, sel, wts);
  k_build<<<1, 256, 0, stream>>>(sel, wts, token_ids, token_wt, counts, offs);

  k_gemm1<<<dim3(kF / 64, kT / 128, kE), 256, 0, stream>>>(
      xb, wv1b, token_ids, counts, offs, act);
  k_gemm2<<<dim3(kD / 64, kT / 128, kE), 256, 0, stream>>>(
      act, w2b, token_ids, token_wt, counts, offs, out);
}
